// SynonymsScaledDotProductAttention_82849919140626
// MI455X (gfx1250) — matrix operand order repaired and checked by reading, not compiled, not run on hardware
//
#include <hip/hip_runtime.h>

typedef __attribute__((ext_vector_type(16))) _Float16 v16h;
typedef __attribute__((ext_vector_type(8)))  _Float16 v8h;
typedef __attribute__((ext_vector_type(8)))  float    v8f;
typedef __attribute__((ext_vector_type(4)))  float    v4f;

#define B_ 8
#define C_ 128
#define Z_ 4
#define S_ 8
#define T_ 1024
#define H_ 64

// Output layout: (scores, context, attn) concatenated flat, all f32.
#define SCORES_OFF 0
#define CTX_OFF    ((size_t)B_ * C_ * Z_ * S_ * T_)          // 33,554,432
#define ATTN_OFF   (CTX_OFF + (size_t)B_ * C_ * Z_ * H_)     // 33,816,576

// 16-lane (half-wave) reductions; xor masks 1,2,4,8 stay inside each
// 16-lane group, matching the WMMA C-matrix layout (lanes 0-15 / 16-31).
// Used exactly once per wave (end of pass 1), not per tile.
__device__ __forceinline__ float grp16_max(float v) {
#pragma unroll
  for (int off = 1; off < 16; off <<= 1)
    v = fmaxf(v, __shfl_xor(v, off, 32));
  return v;
}
__device__ __forceinline__ float grp16_sum(float v) {
#pragma unroll
  for (int off = 1; off < 16; off <<= 1)
    v += __shfl_xor(v, off, 32);
  return v;
}

// ---------------------------------------------------------------------------
// Kernel 0: stage tanh(WK) as f16 (same layout) and tanh(K) as f16
// TRANSPOSED to [bz][h][t], so both GEMMs load B-fragments as two unit-stride
// 16-byte v8h runs per lane along k. One-off 16 MB read / 8 MB write.
// ---------------------------------------------------------------------------
__global__ void __launch_bounds__(256)
tanh_stage_kernel(const float* __restrict__ WK, const float* __restrict__ K,
                  _Float16* __restrict__ wkt, _Float16* __restrict__ ktT) {
  const int n = B_ * Z_ * T_ * H_;
  const int stride = gridDim.x * blockDim.x;
  for (int i = blockIdx.x * blockDim.x + threadIdx.x; i < n; i += stride) {
    wkt[i] = (_Float16)tanhf(WK[i]);
    const int bz = i >> 16;          // / (T_*H_) = 65536
    const int t  = (i >> 6) & (T_ - 1);
    const int h  = i & (H_ - 1);
    ktT[((size_t)bz * H_ + h) * T_ + t] = (_Float16)tanhf(K[i]);
  }
}

// ---------------------------------------------------------------------------
// Kernel 1: scores = A(WQ) x B(tanh(WK)) per (b,z), 16 rows of CS per wave32,
// streaming 64 T-tiles. Per-LANE online softmax (max/sum over this lane's 64
// columns), single cross-lane reduction at the end, then pass 2 emits attn
// from the L2-resident scores it just wrote.
// ---------------------------------------------------------------------------
__global__ void __launch_bounds__(32)
scores_softmax_kernel(const float* __restrict__ WQ,
                      const _Float16* __restrict__ wkt,
                      float* __restrict__ out) {
  const int lane = threadIdx.x;
  const int grp  = lane >> 4;   // 0: lanes 0-15, 1: lanes 16-31
  const int ln   = lane & 15;

  const int bid   = blockIdx.x;
  const int mtile = bid & 63;        // 64 row-tiles of 16 over CS=1024
  const int bz    = bid >> 6;
  const int z     = bz & (Z_ - 1);
  const int b     = bz >> 2;
  const int cs0   = mtile * 16;

  // --- A fragments (WQ raw, f32->f16). 16-bit A layout (ISA 7.12.2):
  // lane<16: K chunks {0..7,16..23}; lane>=16: {8..15,24..31}; row M = ln.
  const int csA = cs0 + ln;
  const int cA = csA >> 3, sA = csA & 7;
  const float* wq = WQ + (((size_t)cA * Z_ + z) * S_ + sA) * H_;

  v16h a[2];
#pragma unroll
  for (int ks = 0; ks < 2; ++ks) {
    const v8f lo = *(const v8f*)(wq + ks * 32 + grp * 8);        // 32B aligned
    const v8f hi = *(const v8f*)(wq + ks * 32 + 16 + grp * 8);
#pragma unroll
    for (int j = 0; j < 8; ++j) {
      a[ks][j]     = (_Float16)lo[j];
      a[ks][8 + j] = (_Float16)hi[j];
    }
  }

  float m[8], l[8];
#pragma unroll
  for (int r = 0; r < 8; ++r) { m[r] = -3.0e38f; l[r] = 0.0f; }

  const _Float16* wkbz = wkt + (((size_t)b * Z_ + z) * T_) * H_;

  for (int jt = 0; jt < T_ / 16; ++jt) {
    // B fragment: column N = t (lane), K=h chunks {0..7,16..23} / {8..15,24..31}.
    const _Float16* wrow = wkbz + (size_t)(jt * 16 + ln) * H_;
    if (jt + 1 < T_ / 16)
      __builtin_prefetch(wrow + 16 * H_, 0, 3);

    v8f acc = {};
#pragma unroll
    for (int ks = 0; ks < 2; ++ks) {
      const v16h bf = __builtin_shufflevector(*(const v8h*)(wrow + ks * 32 + grp * 8), *(const v8h*)(wrow + ks * 32 + 16 + grp * 8), 0, 1, 2, 3, 4, 5, 6, 7, 8, 9, 10, 11, 12, 13, 14, 15);
      acc = __builtin_amdgcn_wmma_f32_16x16x32_f16(false, a[ks], false, bf,
                                                   (short)0, acc, false, false);
    }

    // Store scores tile + per-lane online softmax update.
    // C layout: VGPR r -> M=r (lanes 0-15) / M=8+r (lanes 16-31), N=ln.
#pragma unroll
    for (int r = 0; r < 8; ++r) {
      const int cs = cs0 + grp * 8 + r;
      const int c = cs >> 3, s = cs & 7;
      const size_t idx =
          ((((size_t)b * C_ + c) * Z_ + z) * S_ + s) * T_ + (jt * 16 + ln);
      out[SCORES_OFF + idx] = acc[r];
      const float nm = fmaxf(m[r], acc[r]);
      l[r] = l[r] * __expf(m[r] - nm) + __expf(acc[r] - nm);
      m[r] = nm;
    }
  }

  // One cross-lane reduction per row: row max, rescale lane sums, row sum.
#pragma unroll
  for (int r = 0; r < 8; ++r) {
    const float rm = grp16_max(m[r]);
    const float rs = grp16_sum(l[r] * __expf(m[r] - rm));
    m[r] = rm;
    l[r] = 1.0f / rs;
  }

  // Pass 2: re-read scores (L2-resident: 64 KB/wave, just written) -> attn.
  for (int jt = 0; jt < T_ / 16; ++jt) {
#pragma unroll
    for (int r = 0; r < 8; ++r) {
      const int cs = cs0 + grp * 8 + r;
      const int c = cs >> 3, s = cs & 7;
      const size_t idx =
          ((((size_t)b * C_ + c) * Z_ + z) * S_ + s) * T_ + (jt * 16 + ln);
      out[ATTN_OFF + idx] = __expf(out[SCORES_OFF + idx] - m[r]) * l[r];
    }
  }
}

// ---------------------------------------------------------------------------
// Kernel 2: context = attn @ tanh(K), then mean over s.
// One wave per (b,z, 16-row tile of CS, 16-col tile of H); K-loop over T=1024.
// B-fragments are two unit-stride v8h loads thanks to the transposed ktT staging.
// ---------------------------------------------------------------------------
__global__ void __launch_bounds__(32)
context_kernel(const _Float16* __restrict__ ktT, float* __restrict__ out) {
  const int lane = threadIdx.x;
  const int grp  = lane >> 4;
  const int ln   = lane & 15;

  const int bid   = blockIdx.x;
  const int ntile = bid & 3;          // 4 tiles of 16 over H=64
  const int mtile = (bid >> 2) & 63;  // 64 tiles of 16 over CS=1024
  const int bz    = bid >> 8;
  const int z     = bz & (Z_ - 1);
  const int b     = bz >> 2;

  const int cs0   = mtile * 16;
  const int hbase = ntile * 16;

  const int csA = cs0 + ln;
  const int cA = csA >> 3, sA = csA & 7;
  const float* attn =
      out + ATTN_OFF + ((((size_t)b * C_ + cA) * Z_ + z) * S_ + sA) * T_;
  // B: row h = hbase + ln, contraction t is unit-stride in ktT.
  const _Float16* ktrow = ktT + ((size_t)bz * H_ + hbase + ln) * T_;

  v8f acc = {};
  for (int ktile = 0; ktile < T_ / 32; ++ktile) {
    const int t0 = ktile * 32;
    const v4f a0 = *(const v4f*)(attn + t0 + grp * 8);           // 32B aligned
    const v4f a1 = *(const v4f*)(attn + t0 + grp * 8 + 4);
    const v4f a2 = *(const v4f*)(attn + t0 + 16 + grp * 8);
    const v4f a3 = *(const v4f*)(attn + t0 + 16 + grp * 8 + 4);
    v16h af;
#pragma unroll
    for (int j = 0; j < 4; ++j) {
      af[j]      = (_Float16)a0[j];
      af[4 + j]  = (_Float16)a1[j];
      af[8 + j]  = (_Float16)a2[j];
      af[12 + j] = (_Float16)a3[j];
    }
    const v16h bf = __builtin_shufflevector(*(const v8h*)(ktrow + t0 + grp * 8), *(const v8h*)(ktrow + t0 + 16 + grp * 8), 0, 1, 2, 3, 4, 5, 6, 7, 8, 9, 10, 11, 12, 13, 14, 15);
    acc = __builtin_amdgcn_wmma_f32_16x16x32_f16(false, af, false, bf,
                                                 (short)0, acc, false, false);
  }

  // Rows 0-7 (lane group 0) = c0's 8 synonyms, rows 8-15 (group 1) = c1's.
  // Mean over s == sum of the 8 accumulator VGPRs per lane, * 1/8.
  float v = 0.f;
#pragma unroll
  for (int r = 0; r < 8; ++r) v += acc[r];
  v *= 0.125f;
  const int c = (cs0 >> 3) + grp;
  out[CTX_OFF + (((size_t)b * C_ + c) * Z_ + z) * H_ + hbase + ln] = v;
}

// ---------------------------------------------------------------------------
extern "C" void kernel_launch(void* const* d_in, const int* in_sizes, int n_in,
                              void* d_out, int out_size, void* d_ws,
                              size_t ws_size, hipStream_t stream) {
  (void)in_sizes; (void)n_in; (void)out_size; (void)ws_size;
  const float* WQ = (const float*)d_in[0];
  const float* WK = (const float*)d_in[1];
  const float* K  = (const float*)d_in[2];
  float* out = (float*)d_out;

  // Workspace: tanh(WK) f16 (4 MB) then tanh(K) f16 transposed (4 MB).
  _Float16* wkt = (_Float16*)d_ws;
  _Float16* ktT = wkt + (size_t)B_ * Z_ * T_ * H_;

  tanh_stage_kernel<<<1024, 256, 0, stream>>>(WK, K, wkt, ktT);

  // 32 (b,z) pairs x 64 row-tiles; one wave32 per block.
  scores_softmax_kernel<<<B_ * Z_ * (C_ * S_ / 16), 32, 0, stream>>>(WQ, wkt, out);

  // x4 H-tiles for the context GEMM.
  context_kernel<<<B_ * Z_ * (C_ * S_ / 16) * (H_ / 16), 32, 0, stream>>>(ktT, out);
}
